// CausalSelfAttention_41850161333102
// MI455X (gfx1250) — matrix operand order repaired and checked by reading, not compiled, not run on hardware
//
#include <hip/hip_runtime.h>
#include <cstdint>

// ---------------- Problem constants ----------------
#define BB 2
#define TT 2048
#define CC 1024
#define HH 16
#define DD 64
#define M_TOK (BB * TT)       // 4096 token rows
#define N_QKV (3 * CC)        // 3072
#define K_DIM CC              // 1024

typedef __attribute__((ext_vector_type(16))) _Float16     v16h;
typedef __attribute__((ext_vector_type(8)))  _Float16     v8h;
typedef __attribute__((ext_vector_type(8)))  float        v8f;
typedef __attribute__((ext_vector_type(4)))  unsigned int v4u;
typedef __attribute__((ext_vector_type(8)))  int          v8i;
typedef __attribute__((ext_vector_type(4)))  int          v4i;

#if defined(__has_builtin)
#if __has_builtin(__builtin_amdgcn_tensor_load_to_lds) && \
    __has_builtin(__builtin_amdgcn_s_wait_tensorcnt)
#define USE_TDM 1
#endif
#endif
#ifndef USE_TDM
#define USE_TDM 0
#endif

union Frag16 { v16h v; v8h h[2]; };

__device__ __forceinline__ v8f vzero8() {
    v8f z;
#pragma unroll
    for (int i = 0; i < 8; ++i) z[i] = 0.0f;
    return z;
}

__device__ __forceinline__ v8f wmma_f16(v16h a, v16h b, v8f c) {
    // (neg_a, A, neg_b, B, c_mod, C, reuse_a, reuse_b)
    return __builtin_amdgcn_wmma_f32_16x16x32_f16(false, a, false, b, (short)0, c,
                                                  false, false);
}

// ---------------- fp32 -> fp16 copy ----------------
__global__ void __launch_bounds__(256)
k_f32_to_f16(const float* __restrict__ src, _Float16* __restrict__ dst, int n) {
    int i = (blockIdx.x * 256 + threadIdx.x) * 4;
    if (i + 3 < n) {
        float4 v = *(const float4*)(src + i);
        dst[i + 0] = (_Float16)v.x;
        dst[i + 1] = (_Float16)v.y;
        dst[i + 2] = (_Float16)v.z;
        dst[i + 3] = (_Float16)v.w;
    }
}

// ---------------- fp32 [K][N] -> fp16 [N][K], LDS-tiled 64x64 (coalesced) --------
__global__ void __launch_bounds__(256)
k_transpose_f16(const float* __restrict__ src, _Float16* __restrict__ dst,
                int K, int N) {
    __shared__ _Float16 tile[64][65];
    const int k0 = blockIdx.x * 64;
    const int n0 = blockIdx.y * 64;
    const int tid = threadIdx.x;
#pragma unroll
    for (int i = 0; i < 16; ++i) {
        int idx = tid + i * 256;            // 0..4095
        int r = idx >> 6;                   // local k
        int c = idx & 63;                   // local n  (consecutive per tid)
        tile[c][r] = (_Float16)src[(size_t)(k0 + r) * N + (n0 + c)];
    }
    __syncthreads();
#pragma unroll
    for (int i = 0; i < 16; ++i) {
        int idx = tid + i * 256;
        int rn = idx >> 6;                  // local n
        int ck = idx & 63;                  // local k (consecutive per tid)
        dst[(size_t)(n0 + rn) * K + (k0 + ck)] = tile[rn][ck];
    }
}

// ---------------- WMMA GEMM: C[M][N] = A[M][K] * BT[N][K]^T + bias --------------
// 256 threads = 8 waves arranged 2(M) x 4(N); wave tile 32x64 (2x4 WMMA tiles);
// block tile 64x256; K step 32.
__global__ void __launch_bounds__(256)
k_wmma_gemm(const _Float16* __restrict__ A, const _Float16* __restrict__ BT,
            const float* __restrict__ bias, void* __restrict__ out,
            int M, int N, int K, int out_is_f16) {
    const int lane = threadIdx.x & 31;
    const int w    = threadIdx.x >> 5;
    const int lm   = lane & 15;
    const int lh   = lane >> 4;
    const int wm   = w >> 2;                // 0..1
    const int wn   = w & 3;                 // 0..3
    const int m0   = blockIdx.x * 64 + wm * 32;
    const int n0   = blockIdx.y * 256 + wn * 64;

    v8f acc[2][4];
#pragma unroll
    for (int mt = 0; mt < 2; ++mt)
#pragma unroll
        for (int nt = 0; nt < 4; ++nt) acc[mt][nt] = vzero8();

    for (int k0 = 0; k0 < K; k0 += 32) {
        Frag16 af[2], bf[4];
#pragma unroll
        for (int mt = 0; mt < 2; ++mt) {
            const _Float16* ap = A + (size_t)(m0 + mt * 16 + lm) * K + k0 + lh * 16;
            af[mt].h[0] = *(const v8h*)ap;          // k = base+0..7
            af[mt].h[1] = *(const v8h*)(ap + 8);    // k = base+8..15
        }
#pragma unroll
        for (int nt = 0; nt < 4; ++nt) {
            const _Float16* bp = BT + (size_t)(n0 + nt * 16 + lm) * K + k0 + lh * 16;
            bf[nt].v = *(const v16h*)bp;            // 16 contiguous k
        }
#pragma unroll
        for (int mt = 0; mt < 2; ++mt)
#pragma unroll
            for (int nt = 0; nt < 4; ++nt)
                acc[mt][nt] = wmma_f16(af[mt].v, bf[nt].v, acc[mt][nt]);
    }

    float bs[4];
#pragma unroll
    for (int nt = 0; nt < 4; ++nt) bs[nt] = bias[n0 + nt * 16 + lm];

#pragma unroll
    for (int mt = 0; mt < 2; ++mt) {
#pragma unroll
        for (int nt = 0; nt < 4; ++nt) {
            const int col = n0 + nt * 16 + lm;
#pragma unroll
            for (int j = 0; j < 8; ++j) {
                const int row = m0 + mt * 16 + j + 8 * lh;
                float v = acc[mt][nt][j] + bs[nt];
                if (out_is_f16)
                    ((_Float16*)out)[(size_t)row * N + col] = (_Float16)v;
                else
                    ((float*)out)[(size_t)row * N + col] = v;
            }
        }
    }
}

// ---------------- Flash attention (causal, online softmax) ----------------------
// Grid: B*H*(T/64) blocks, 128 threads (4 waves). Wave owns 16 queries.
// Scores computed transposed (S^T = K * Q^T) so each lane owns one query column.
// K tile staged to LDS once per block via the Tensor Data Mover; V tile staged
// transposed by VALU; P round-trips through per-wave LDS to become a B-fragment.
__global__ void __launch_bounds__(128)
k_flash_attn(const _Float16* __restrict__ qkv, _Float16* __restrict__ yout) {
    const int per_bh = TT / 64;
    const int bh   = blockIdx.x / per_bh;
    const int qblk = blockIdx.x % per_bh;
    const int b = bh / HH, h = bh % HH;

    const int lane = threadIdx.x & 31;
    const int w    = threadIdx.x >> 5;
    const int lm   = lane & 15;
    const int lh   = lane >> 4;
    const int q0   = qblk * 64 + w * 16;

    const size_t base = (size_t)b * TT * N_QKV + (size_t)h * DD;
    const _Float16* Qb = qkv + base;
    const _Float16* Kb = qkv + base + CC;
    const _Float16* Vb = qkv + base + 2 * CC;

    __shared__ _Float16 Ks[32][64];         // K tile, row-major (= A-frag layout)
    __shared__ _Float16 Vt[DD][40];         // V tile transposed, padded rows
    __shared__ _Float16 Pl[4][16][32];      // per-wave P, q-major

    // Q^T B-fragments, pre-scaled by 1/sqrt(D)
    Frag16 bq0, bq1;
    {
        const _Float16* qp = Qb + (size_t)(q0 + lm) * N_QKV + lh * 16;
        bq0.v = *(const v16h*)qp;
        bq1.v = *(const v16h*)(qp + 32);
#pragma unroll
        for (int i = 0; i < 16; ++i) {
            bq0.v[i] = bq0.v[i] * (_Float16)0.125f;
            bq1.v[i] = bq1.v[i] * (_Float16)0.125f;
        }
    }

    float m_run = -1.0e30f, l_run = 0.0f;
    v8f Y[4];
#pragma unroll
    for (int dt = 0; dt < 4; ++dt) Y[dt] = vzero8();

#if USE_TDM
    const unsigned lds_k = (unsigned)(uintptr_t)(&Ks[0][0]);
#endif

    const int nkb = qblk * 2 + 2;           // 32-key blocks covering causal span
    for (int kb = 0; kb < nkb; ++kb) {
        // ---- stage K tile [32 keys][64 d] row-major into LDS ----
#if USE_TDM
        if (w == 0) {
            const _Float16* ktile = Kb + (size_t)(kb * 32) * N_QKV;
            unsigned long long ga = (unsigned long long)(uintptr_t)ktile;
            v4u g0;
            g0[0] = 1u;                                          // count=1
            g0[1] = lds_k;                                       // lds_addr
            g0[2] = (unsigned)(ga & 0xFFFFFFFFu);                // global_addr lo
            g0[3] = (unsigned)((ga >> 32) & 0x01FFFFFFu) | (2u << 30); // hi + type=2
            v8i g1;
            g1[0] = (1 << 16);                                   // data_size = 2B
            g1[1] = (3072 << 16);                                // tensor_dim0 lo16
            g1[2] = (2048 << 16);                                // dim0 hi=0, dim1 lo16
            g1[3] = (64 << 16);                                  // dim1 hi=0, tile_dim0=64
            g1[4] = 32;                                          // tile_dim1=32, tile_dim2=0
            g1[5] = 3072;                                        // tensor_dim0_stride lo32
            g1[6] = 0;
            g1[7] = 0;
            v4i z4 = {0, 0, 0, 0};
            v8i z8 = {0, 0, 0, 0, 0, 0, 0, 0};
            __builtin_amdgcn_tensor_load_to_lds(g0, g1, z4, z4, z8, 0);
        }
#else
        {
            const int kl = threadIdx.x >> 2;         // 0..31
            const int db = (threadIdx.x & 3) * 16;   // d base
            const _Float16* kp = Kb + (size_t)(kb * 32 + kl) * N_QKV + db;
            *(v16h*)&Ks[kl][db] = *(const v16h*)kp;
        }
#endif
        // ---- stage V tile transposed: V[32 keys][64 d] -> Vt[d][key] ----
        {
            const int kl = threadIdx.x >> 2;
            const int db = (threadIdx.x & 3) * 16;
            const _Float16* vp = Vb + (size_t)(kb * 32 + kl) * N_QKV + db;
            v16h vv = *(const v16h*)vp;
#pragma unroll
            for (int i = 0; i < 16; ++i) Vt[db + i][kl] = vv[i];
        }
#if USE_TDM
        if (w == 0) __builtin_amdgcn_s_wait_tensorcnt((short)0);
#endif
        __syncthreads();

        if (kb * 32 <= q0 + 15) {               // wave-uniform causal skip
            const int q = q0 + lm;
            float sv[16];
            float tmax = -1.0e30f;
#pragma unroll
            for (int hf = 0; hf < 2; ++hf) {
                const int keyl = hf * 16 + lm;
                Frag16 a0, a1;
                a0.h[0] = *(const v8h*)&Ks[keyl][lh * 16];       // d 0..31
                a0.h[1] = *(const v8h*)&Ks[keyl][lh * 16 + 8];
                a1.h[0] = *(const v8h*)&Ks[keyl][lh * 16 + 32];  // d 32..63
                a1.h[1] = *(const v8h*)&Ks[keyl][lh * 16 + 40];
                v8f s = vzero8();
                s = wmma_f16(a0.v, bq0.v, s);
                s = wmma_f16(a1.v, bq1.v, s);
#pragma unroll
                for (int j = 0; j < 8; ++j) {
                    const int ky = kb * 32 + hf * 16 + j + 8 * lh;
                    float sval = (ky <= q) ? s[j] : -1.0e30f;
                    sv[hf * 8 + j] = sval;
                    tmax = fmaxf(tmax, sval);
                }
            }
            tmax = fmaxf(tmax, __shfl_xor(tmax, 16));
            const float mnew  = fmaxf(m_run, tmax);
            const float alpha = __expf(m_run - mnew);
            float psum = 0.0f;
#pragma unroll
            for (int hf = 0; hf < 2; ++hf)
#pragma unroll
                for (int j = 0; j < 8; ++j) {
                    float p = __expf(sv[hf * 8 + j] - mnew);
                    psum += p;
                    Pl[w][lm][hf * 16 + j + 8 * lh] = (_Float16)p;
                }
            psum += __shfl_xor(psum, 16);
            l_run = l_run * alpha + psum;
            m_run = mnew;
#pragma unroll
            for (int dt = 0; dt < 4; ++dt)
#pragma unroll
                for (int j = 0; j < 8; ++j) Y[dt][j] *= alpha;

            Frag16 bp;                                   // P^T B-fragment
            bp.v = *(const v16h*)&Pl[w][lm][lh * 16];
#pragma unroll
            for (int dt = 0; dt < 4; ++dt) {
                Frag16 av;                               // V^T A-fragment
                const int d = dt * 16 + lm;
                av.h[0] = *(const v8h*)&Vt[d][lh * 16];
                av.h[1] = *(const v8h*)&Vt[d][lh * 16 + 8];
                Y[dt] = wmma_f16(av.v, bp.v, Y[dt]);
            }
        }
        __syncthreads();
    }

    const float inv = 1.0f / l_run;
    const int t = q0 + lm;
    _Float16* yp = yout + ((size_t)(b * TT + t)) * CC + h * DD;
#pragma unroll
    for (int dt = 0; dt < 4; ++dt)
#pragma unroll
        for (int j = 0; j < 8; ++j)
            yp[dt * 16 + j + 8 * lh] = (_Float16)(Y[dt][j] * inv);
}

// ---------------- Host launch ----------------
extern "C" void kernel_launch(void* const* d_in, const int* in_sizes, int n_in,
                              void* d_out, int out_size, void* d_ws, size_t ws_size,
                              hipStream_t stream) {
    (void)in_sizes; (void)n_in; (void)out_size; (void)ws_size;

    const float* x      = (const float*)d_in[0];   // [B,T,C]
    const float* w_attn = (const float*)d_in[1];   // [C,3C]
    const float* b_attn = (const float*)d_in[2];   // [3C]
    const float* w_proj = (const float*)d_in[3];   // [C,C]
    const float* b_proj = (const float*)d_in[4];   // [C]

    char* ws = (char*)d_ws;
    _Float16* xh   = (_Float16*)(ws);                          //  8 MB [4096][1024]
    _Float16* waT  = (_Float16*)(ws + (8ull  << 20));          //  6 MB [3072][1024]
    _Float16* wpT  = (_Float16*)(ws + (14ull << 20));          //  2 MB [1024][1024]
    _Float16* qkvh = (_Float16*)(ws + (16ull << 20));          // 24 MB [4096][3072]
    _Float16* yh   = (_Float16*)(ws + (40ull << 20));          //  8 MB [4096][1024]

    // 1) precision conversion / coalesced weight transposition
    k_f32_to_f16<<<M_TOK * K_DIM / 1024, 256, 0, stream>>>(x, xh, M_TOK * K_DIM);
    {
        dim3 g(K_DIM / 64, N_QKV / 64);
        k_transpose_f16<<<g, 256, 0, stream>>>(w_attn, waT, K_DIM, N_QKV);
    }
    {
        dim3 g(K_DIM / 64, CC / 64);
        k_transpose_f16<<<g, 256, 0, stream>>>(w_proj, wpT, K_DIM, CC);
    }

    // 2) QKV projection: [4096,1024] x [1024,3072] + bias -> fp16
    {
        dim3 grid(M_TOK / 64, N_QKV / 256);
        k_wmma_gemm<<<grid, 256, 0, stream>>>(xh, waT, b_attn, (void*)qkvh,
                                              M_TOK, N_QKV, K_DIM, 1);
    }

    // 3) causal flash attention -> yh fp16 [4096][1024]
    k_flash_attn<<<BB * HH * (TT / 64), 128, 0, stream>>>(qkvh, yh);

    // 4) output projection: [4096,1024] x [1024,1024] + bias -> fp32 d_out
    {
        dim3 grid(M_TOK / 64, CC / 256);
        k_wmma_gemm<<<grid, 256, 0, stream>>>(yh, wpT, b_proj, d_out,
                                              M_TOK, CC, K_DIM, 0);
    }
}
